// GraphModule_59012850647688
// MI455X (gfx1250) — matrix operand order repaired and checked by reading, not compiled, not run on hardware
//
#include <hip/hip_runtime.h>

// ---------------- types for WMMA ----------------
typedef unsigned short u16;
typedef __attribute__((ext_vector_type(16))) __bf16 bf16x16;
typedef __attribute__((ext_vector_type(16))) u16    u16x16;
typedef __attribute__((ext_vector_type(8)))  float  f32x8;
typedef __attribute__((ext_vector_type(4)))  int    v4i;

#define IN_DIM 128
#define DD     256
#define NH     4
#define CC     64
#define NEG_SLOPE 0.2f

// async global->LDS availability (gfx1250)
#if defined(__has_builtin)
#  if __has_builtin(__builtin_amdgcn_global_load_async_to_lds_b128)
#    define HAVE_ASYNC_LDS 1
#  endif
#endif
#ifndef HAVE_ASYNC_LDS
#  define HAVE_ASYNC_LDS 0
#endif

// ---------------- helpers ----------------
__device__ __forceinline__ u16 f2bf(float f) {
    unsigned b = __float_as_uint(f);
    unsigned r = b + 0x7FFFu + ((b >> 16) & 1u);   // round-to-nearest-even
    return (u16)(r >> 16);
}
__device__ __forceinline__ float bf2f(u16 h) {
    return __uint_as_float(((unsigned)h) << 16);
}
// monotone float <-> ordered-uint map for atomicMax on floats
__device__ __forceinline__ unsigned f2o(float f) {
    unsigned b = __float_as_uint(f);
    return (b & 0x80000000u) ? ~b : (b | 0x80000000u);
}
__device__ __forceinline__ float o2f(unsigned u) {
    unsigned b = (u & 0x80000000u) ? (u & 0x7FFFFFFFu) : ~u;
    return __uint_as_float(b);
}
__device__ __forceinline__ float lrelu(float x) {
    return x > 0.f ? x : NEG_SLOPE * x;
}

// ---------------- per-layer W split prepass: fp32 -> (hi, lo) bf16 ----------------
__global__ void split_k(const float* __restrict__ src, u16* __restrict__ hi,
                        u16* __restrict__ lo, int n)
{
    int i = blockIdx.x * blockDim.x + threadIdx.x;
    if (i >= n) return;
    float v = src[i];
    u16 h = f2bf(v);
    hi[i] = h;
    lo[i] = f2bf(v - bf2f(h));
}

// ---------------- GEMM: H = X @ W^T via split-bf16 WMMA ----------------
// X: [Nn, Kdim] fp32, Whi/Wlo: [256, Kdim] bf16 bits, H: [Nn, 256] fp32
#define TM 64
#define TN 64
#define KT 32
#define BSTR (KT + 8)    // B row stride: 40 u16 = 80B, 16B aligned, bank-friendly

__global__ __launch_bounds__(128) void gemm_wmma_k(
    const float* __restrict__ X, const u16* __restrict__ Whi, const u16* __restrict__ Wlo,
    float* __restrict__ Hout, int Nn, int Kdim)
{
    __shared__ u16 Ahi[TM][KT + 2], Alo[TM][KT + 2];
    __shared__ u16 Bhi[TN][BSTR],   Blo[TN][BSTR];

    const int tid  = threadIdx.x;
    const int wave = tid >> 5;
    const int lane = tid & 31;
    const int half = lane >> 4;
    const int l16  = lane & 15;
    const int m0   = blockIdx.y * TM;
    const int n0   = blockIdx.x * TN;

    f32x8 acc[4];
#pragma unroll
    for (int t = 0; t < 4; ++t) acc[t] = (f32x8){0.f,0.f,0.f,0.f,0.f,0.f,0.f,0.f};

    for (int kb = 0; kb < Kdim; kb += KT) {
        // ---- stage B tiles: raw 16B copies of pre-split W (async if available) ----
        for (int i = tid; i < TN * 4; i += 128) {          // 4 x 16B chunks per row
            int row = i >> 2, ch = (i & 3) * 8;            // ch in u16 units
            const u16* gh = Whi + (size_t)(n0 + row) * Kdim + kb + ch;
            const u16* gl = Wlo + (size_t)(n0 + row) * Kdim + kb + ch;
#if HAVE_ASYNC_LDS
            __builtin_amdgcn_global_load_async_to_lds_b128(
                (__attribute__((address_space(1))) v4i*)gh,
                (__attribute__((address_space(3))) v4i*)&Bhi[row][ch], 0, 0);
            __builtin_amdgcn_global_load_async_to_lds_b128(
                (__attribute__((address_space(1))) v4i*)gl,
                (__attribute__((address_space(3))) v4i*)&Blo[row][ch], 0, 0);
#else
            *(uint4*)&Bhi[row][ch] = *(const uint4*)gh;
            *(uint4*)&Blo[row][ch] = *(const uint4*)gl;
#endif
        }
        // ---- stage A tile: load fp32 X, split to hi/lo bf16 in LDS ----
        for (int i = tid; i < TM * KT; i += 128) {
            int r = i >> 5, c = i & 31;
            int gm = m0 + r;
            float v = (gm < Nn) ? X[(size_t)gm * Kdim + kb + c] : 0.f;
            u16 vh = f2bf(v);
            Ahi[r][c] = vh;
            Alo[r][c] = f2bf(v - bf2f(vh));
        }
#if HAVE_ASYNC_LDS
#  if __has_builtin(__builtin_amdgcn_s_wait_asynccnt)
        __builtin_amdgcn_s_wait_asynccnt(0);
#  else
        asm volatile("s_wait_asynccnt 0" ::: "memory");
#  endif
#endif
        __syncthreads();

        // A fragment: 16x32 bf16, CDNA5 lane layout
        const int mrow = wave * 16 + l16;
        u16x16 ahu, alu;
#pragma unroll
        for (int j = 0; j < 16; ++j) {
            int v = j >> 1, p = j & 1;
            int kl = (v < 4 ? 2 * v : 8 + 2 * v) + 8 * half + p;
            ahu[j] = Ahi[mrow][kl];
            alu[j] = Alo[mrow][kl];
        }
        bf16x16 a_hi = __builtin_bit_cast(bf16x16, ahu);
        bf16x16 a_lo = __builtin_bit_cast(bf16x16, alu);

#pragma unroll
        for (int t = 0; t < 4; ++t) {
            const int ncol = t * 16 + l16;
            u16x16 bhu, blu;
#pragma unroll
            for (int j = 0; j < 16; ++j) {
                int kl = (j < 8 ? j : j + 8) + 8 * half;       // B 32x16 lane layout
                bhu[j] = Bhi[ncol][kl];
                blu[j] = Blo[ncol][kl];
            }
            bf16x16 b_hi = __builtin_bit_cast(bf16x16, bhu);
            bf16x16 b_lo = __builtin_bit_cast(bf16x16, blu);
            acc[t] = __builtin_amdgcn_wmma_f32_16x16x32_bf16(false, a_hi, false, b_hi, (short)0, acc[t], false, false);
            acc[t] = __builtin_amdgcn_wmma_f32_16x16x32_bf16(false, a_hi, false, b_lo, (short)0, acc[t], false, false);
            acc[t] = __builtin_amdgcn_wmma_f32_16x16x32_bf16(false, a_lo, false, b_hi, (short)0, acc[t], false, false);
        }
        __syncthreads();
    }

    // store C/D: VGPR r -> M = r + 8*half, col = l16
#pragma unroll
    for (int t = 0; t < 4; ++t) {
        int n = n0 + t * 16 + l16;
#pragma unroll
        for (int r = 0; r < 8; ++r) {
            int m = m0 + wave * 16 + 8 * half + r;
            if (m < Nn) Hout[(size_t)m * DD + n] = acc[t][r];
        }
    }
}

// ---------------- per-node attention logits ----------------
__global__ void alpha_k(const float* __restrict__ H,
                        const float* __restrict__ att_s, const float* __restrict__ att_d,
                        float* __restrict__ As, float* __restrict__ Ad, int Nn)
{
    int i = blockIdx.x * blockDim.x + threadIdx.x;
    if (i >= Nn * NH) return;
    int n = i >> 2, h = i & 3;
    const float* hp = H + (size_t)n * DD + h * CC;
    const float* asx = att_s + h * CC;
    const float* adx = att_d + h * CC;
    float ss = 0.f, sd = 0.f;
#pragma unroll 8
    for (int c = 0; c < CC; ++c) {
        float v = hp[c];
        ss += v * asx[c];
        sd += v * adx[c];
    }
    As[i] = ss;
    Ad[i] = sd;
}

// ---------------- init: zero out, reset segment max/sum ----------------
__global__ void init_k(float* __restrict__ Out, unsigned* __restrict__ Mord,
                       float* __restrict__ Ssum, int Nn)
{
    int i = blockIdx.x * blockDim.x + threadIdx.x;
    if (i < Nn * DD) Out[i] = 0.f;
    if (i < Nn * NH) { Mord[i] = 0u; Ssum[i] = 1e-16f; }
}

// ---------------- edge pass 1: segment max ----------------
__global__ void edge_max_k(const int* __restrict__ src, const int* __restrict__ dst,
                           int E, int Nn, const float* __restrict__ As,
                           const float* __restrict__ Ad, unsigned* __restrict__ Mord)
{
    int e = blockIdx.x * blockDim.x + threadIdx.x;
    if (e >= E + Nn) return;
    int s, d;
    if (e < E) { s = src[e]; d = dst[e]; } else { s = d = e - E; }
#pragma unroll
    for (int h = 0; h < NH; ++h) {
        float a = lrelu(As[s * NH + h] + Ad[d * NH + h]);
        atomicMax(&Mord[d * NH + h], f2o(a));
    }
}

// ---------------- edge pass 2: segment sum of exp ----------------
__global__ void edge_sum_k(const int* __restrict__ src, const int* __restrict__ dst,
                           int E, int Nn, const float* __restrict__ As,
                           const float* __restrict__ Ad, const unsigned* __restrict__ Mord,
                           float* __restrict__ Ssum)
{
    int e = blockIdx.x * blockDim.x + threadIdx.x;
    if (e >= E + Nn) return;
    int s, d;
    if (e < E) { s = src[e]; d = dst[e]; } else { s = d = e - E; }
#pragma unroll
    for (int h = 0; h < NH; ++h) {
        float a = lrelu(As[s * NH + h] + Ad[d * NH + h]);
        float m = o2f(Mord[d * NH + h]);
        atomicAdd(&Ssum[d * NH + h], expf(a - m));
    }
}

// ---------------- edge pass 3: weighted scatter-add (wave per edge) ----------------
__global__ __launch_bounds__(256) void edge_agg_k(
    const int* __restrict__ src, const int* __restrict__ dst,
    int E, int Nn, const float* __restrict__ As, const float* __restrict__ Ad,
    const unsigned* __restrict__ Mord, const float* __restrict__ Ssum,
    const float* __restrict__ H, float* __restrict__ Out)
{
    int e = blockIdx.x * 8 + (threadIdx.x >> 5);
    if (e >= E + Nn) return;
    int lane = threadIdx.x & 31;
    int s, d;
    if (e < E) { s = src[e]; d = dst[e]; } else { s = d = e - E; }

    int h = lane >> 3;                       // 8 lanes per head, 8 floats per lane
    float a = lrelu(As[s * NH + h] + Ad[d * NH + h]);
    float m = o2f(Mord[d * NH + h]);
    float coef = expf(a - m) / Ssum[d * NH + h];

    const float4* hp = (const float4*)(H + (size_t)s * DD + lane * 8);
    float4 v0 = hp[0];
    float4 v1 = hp[1];
    float* op = Out + (size_t)d * DD + lane * 8;
    atomicAdd(op + 0, coef * v0.x);
    atomicAdd(op + 1, coef * v0.y);
    atomicAdd(op + 2, coef * v0.z);
    atomicAdd(op + 3, coef * v0.w);
    atomicAdd(op + 4, coef * v1.x);
    atomicAdd(op + 5, coef * v1.y);
    atomicAdd(op + 6, coef * v1.z);
    atomicAdd(op + 7, coef * v1.w);
}

// ---------------- bias + (optional) relu ----------------
__global__ void bias_relu_k(float* __restrict__ Out, const float* __restrict__ bias,
                            int Nn, int do_relu)
{
    int i = blockIdx.x * blockDim.x + threadIdx.x;
    if (i >= Nn * DD) return;
    float v = Out[i] + bias[i & (DD - 1)];
    Out[i] = do_relu ? fmaxf(v, 0.f) : v;
}

// ---------------- host orchestration ----------------
extern "C" void kernel_launch(void* const* d_in, const int* in_sizes, int n_in,
                              void* d_out, int out_size, void* d_ws, size_t ws_size,
                              hipStream_t stream)
{
    const float* x_in = (const float*)d_in[0];
    const int*   ei   = (const int*)d_in[1];
    const int Nn = in_sizes[0] / IN_DIM;        // 50000
    const int E  = in_sizes[1] / 2;             // 400000
    const int* src = ei;
    const int* dst = ei + E;

    float* Out = (float*)d_out;                 // ping buffer + final output
    float* ws  = (float*)d_ws;
    float* Hbuf = ws;                           // [Nn, 256] projected features
    float* As   = Hbuf + (size_t)Nn * DD;       // [Nn, 4]
    float* Ad   = As + (size_t)Nn * NH;         // [Nn, 4]
    unsigned* Mord = (unsigned*)(Ad + (size_t)Nn * NH);  // [Nn, 4]
    float* Ssum = (float*)(Mord + (size_t)Nn * NH);      // [Nn, 4]
    u16* Whi = (u16*)(Ssum + (size_t)Nn * NH);  // [256, 256] bf16 bits (max)
    u16* Wlo = Whi + (size_t)DD * DD;

    const int ET = E + Nn;                      // edges incl. self-loops

    for (int l = 0; l < 5; ++l) {
        const float* Wl   = (const float*)d_in[2 + 4 * l];
        const float* a_s  = (const float*)d_in[3 + 4 * l];
        const float* a_d  = (const float*)d_in[4 + 4 * l];
        const float* bias = (const float*)d_in[5 + 4 * l];
        const int Kdim = (l == 0) ? IN_DIM : DD;
        const float* X = (l == 0) ? x_in : (const float*)Out;

        // 0) split W into hi/lo bf16 once per layer
        const int wcount = DD * Kdim;
        hipLaunchKernelGGL(split_k, dim3((wcount + 255) / 256), dim3(256), 0, stream,
                           Wl, Whi, Wlo, wcount);

        // 1) projection (reads X; after this X's buffer is dead -> reuse as Out)
        dim3 ggrid(DD / TN, (Nn + TM - 1) / TM);
        hipLaunchKernelGGL(gemm_wmma_k, ggrid, dim3(128), 0, stream, X, Whi, Wlo, Hbuf, Nn, Kdim);

        // 2) per-node logits
        hipLaunchKernelGGL(alpha_k, dim3((Nn * NH + 255) / 256), dim3(256), 0, stream,
                           Hbuf, a_s, a_d, As, Ad, Nn);

        // 3) reset Out / segment state
        hipLaunchKernelGGL(init_k, dim3((Nn * DD + 255) / 256), dim3(256), 0, stream,
                           Out, Mord, Ssum, Nn);

        // 4) segment softmax + aggregation
        hipLaunchKernelGGL(edge_max_k, dim3((ET + 255) / 256), dim3(256), 0, stream,
                           src, dst, E, Nn, As, Ad, Mord);
        hipLaunchKernelGGL(edge_sum_k, dim3((ET + 255) / 256), dim3(256), 0, stream,
                           src, dst, E, Nn, As, Ad, Mord, Ssum);
        hipLaunchKernelGGL(edge_agg_k, dim3((ET + 7) / 8), dim3(256), 0, stream,
                           src, dst, E, Nn, As, Ad, Mord, Ssum, Hbuf, Out);

        // 5) bias + relu (no relu on last layer)
        hipLaunchKernelGGL(bias_relu_k, dim3((Nn * DD + 255) / 256), dim3(256), 0, stream,
                           Out, bias, Nn, (l < 4) ? 1 : 0);
    }
}
